// MMDLoss_65798898974810
// MI455X (gfx1250) — matrix operand order repaired and checked by reading, not compiled, not run on hardware
//
#include <hip/hip_runtime.h>

typedef __attribute__((ext_vector_type(8)))  float  v8f;
typedef __attribute__((ext_vector_type(8)))  __bf16 v8bf;
typedef __attribute__((ext_vector_type(16))) __bf16 v16bf;

#define N_TOT   8192
#define DIMS    256
#define HALF_B  4096
#define BLK     128
#define BK      32
#define LDS_STR 40   // 32 + 8 bf16 pad: 80B rows -> 16B aligned, conflict-free bank starts

// ---------------- kernel 0: zero accumulators (ws is poisoned by harness) ----
__global__ void mmd_init_kernel(float* __restrict__ ws) {
    int t = threadIdx.x;
    if (t < 8) ws[t] = 0.0f;      // [0]=sumsq, [1]=signed acc, [2..6]=cvals
    ws[8 + t] = 0.0f;             // colsum[256]
}

__device__ __forceinline__ const float* row_ptr(const float* __restrict__ s,
                                                const float* __restrict__ t, int r) {
    return (r < HALF_B) ? (s + (size_t)r * DIMS) : (t + (size_t)(r - HALF_B) * DIMS);
}

// ------- kernel 1: row sum-of-squares + column sums + f32 -> bf16 hi/lo -----
__global__ __launch_bounds__(256) void mmd_stats_kernel(
    const float* __restrict__ src, const float* __restrict__ tgt,
    float* __restrict__ sq, float* __restrict__ colsum, float* __restrict__ sumsq_acc,
    __bf16* __restrict__ thi, __bf16* __restrict__ tlo) {
    const int tid  = threadIdx.x;
    const int lane = tid & 31;
    const int wid  = tid >> 5;
    const int rowbase = blockIdx.x * 256;

    // Phase A: one wave per row (32 rows per wave); also emit split-bf16 copy
    float wave_sumsq = 0.0f;
    for (int r = 0; r < 32; ++r) {
        int row = rowbase + wid * 32 + r;
        const float* p = row_ptr(src, tgt, row) + lane * 8;
        float x[8];
        *(float4*)&x[0] = *(const float4*)p;
        *(float4*)&x[4] = *(const float4*)(p + 4);
        v8bf hv, lv;
        float s = 0.0f;
        #pragma unroll
        for (int e = 0; e < 8; ++e) {
            s += x[e] * x[e];
            __bf16 h = (__bf16)x[e];          // RNE f32->bf16
            hv[e] = h;
            lv[e] = (__bf16)(x[e] - (float)h); // residual
        }
        size_t base = (size_t)row * DIMS + lane * 8;
        *(v8bf*)(thi + base) = hv;
        *(v8bf*)(tlo + base) = lv;
        #pragma unroll
        for (int off = 16; off > 0; off >>= 1) s += __shfl_xor(s, off, 32);
        if (lane == 0) { sq[row] = s; wave_sumsq += s; }
    }
    if (lane == 0) atomicAdd(sumsq_acc, wave_sumsq);

    // Phase B: partial column sums (thread == column, coalesced rows)
    float cs = 0.0f;
    for (int r = 0; r < 256; ++r) cs += row_ptr(src, tgt, rowbase + r)[tid];
    atomicAdd(&colsum[tid], cs);
}

// ---------------- kernel 2: bandwidth -> per-scale coefficients -------------
__global__ void mmd_bw_kernel(const float* __restrict__ sumsq_acc,
                              const float* __restrict__ colsum,
                              float* __restrict__ cvals) {
    float sc = 0.0f;
    for (int d = 0; d < DIMS; ++d) { float c = colsum[d]; sc += c * c; }
    const float n = (float)N_TOT;
    // sum(L2) = 2n*sum(sq) - 2*||colsum||^2 (clamp only affects diag rounding noise)
    float sumL2 = 2.0f * n * sumsq_acc[0] - 2.0f * sc;
    float bw = sumL2 / (n * n - n + 1e-8f);
    bw *= 0.25f;                               // / KERNEL_MUL^(KERNEL_NUM//2)
    float m = 1.0f;
    for (int i = 0; i < 5; ++i) { cvals[i] = 1.0f / (bw * m + 1e-8f); m *= 2.0f; }
}

// ---- A fragment (16x32 bf16): lanes 0-15 K=0-15, lanes 16-31 K=16-31
__device__ __forceinline__ v16bf load_fragA(const __bf16* __restrict__ panel,
                                            int row, int colA) {
    const __bf16* p = panel + (size_t)row * LDS_STR + colA;
    v8bf a0 = *(const v8bf*)p;
    v8bf a1 = *(const v8bf*)(p + 8);
    return __builtin_shufflevector(a0, a1, 0,1,2,3,4,5,6,7,8,9,10,11,12,13,14,15);
}
// ---- B fragment (32x16 bf16): lanes 0-15 K=0-15, lanes 16-31 K=16-31
__device__ __forceinline__ v16bf load_fragB(const __bf16* __restrict__ panel,
                                            int row, int colB) {
    const __bf16* p = panel + (size_t)row * LDS_STR + colB;
    v8bf b0 = *(const v8bf*)p;
    v8bf b1 = *(const v8bf*)(p + 8);
    return __builtin_shufflevector(b0, b1, 0,1,2,3,4,5,6,7,8,9,10,11,12,13,14,15);
}

// ------- kernel 3: split-bf16 (hi+lo) WMMA gram + gaussian kernel epilogue --
__global__ __launch_bounds__(256) void mmd_gram_kernel(
    const __bf16* __restrict__ thi, const __bf16* __restrict__ tlo,
    const float* __restrict__ sq, const float* __restrict__ cvals,
    float* __restrict__ signed_acc) {
    __shared__ __align__(16) __bf16 AsH[BLK][LDS_STR];
    __shared__ __align__(16) __bf16 AsL[BLK][LDS_STR];
    __shared__ __align__(16) __bf16 BsH[BLK][LDS_STR];
    __shared__ __align__(16) __bf16 BsL[BLK][LDS_STR];
    __shared__ float blk_acc;

    const int tid  = threadIdx.x;
    const int lane = tid & 31;
    const int wid  = tid >> 5;
    const int wrow = wid & 3;     // 4 wave-rows  -> 32 rows each
    const int wcol = wid >> 2;    // 2 wave-cols  -> 64 cols each
    const int i0 = blockIdx.y * BLK;
    const int j0 = blockIdx.x * BLK;

    if (tid == 0) blk_acc = 0.0f;

    v8f acc[2][4];
    #pragma unroll
    for (int tm = 0; tm < 2; ++tm)
        #pragma unroll
        for (int tn = 0; tn < 4; ++tn) acc[tm][tn] = (v8f){};

    const int lm   = lane & 15;
    const int colA = (lane < 16) ? 0 : 16;   // + second b128 at +8
    const int colB = (lane < 16) ? 0 : 16;   // + second b128 at +8

    for (int kk = 0; kk < DIMS; kk += BK) {
        __syncthreads();
        // stage 4 panels of 128x32 bf16 (each: 512 x 16B, 2 chunks/thread)
        #pragma unroll
        for (int l = 0; l < 2; ++l) {
            int idx = l * 256 + tid;
            int r  = idx >> 2;
            int c8 = (idx & 3) << 3;
            size_t ga = (size_t)(i0 + r) * DIMS + kk + c8;
            size_t gb = (size_t)(j0 + r) * DIMS + kk + c8;
            *(uint4*)&AsH[r][c8] = *(const uint4*)(thi + ga);
            *(uint4*)&AsL[r][c8] = *(const uint4*)(tlo + ga);
            *(uint4*)&BsH[r][c8] = *(const uint4*)(thi + gb);
            *(uint4*)&BsL[r][c8] = *(const uint4*)(tlo + gb);
        }
        __syncthreads();

        v16bf ah[2], al[2], b[4];
        #pragma unroll
        for (int tm = 0; tm < 2; ++tm)
            ah[tm] = load_fragA(&AsH[0][0], wrow * 32 + tm * 16 + lm, colA);
        #pragma unroll
        for (int tn = 0; tn < 4; ++tn)
            b[tn] = load_fragB(&BsH[0][0], wcol * 64 + tn * 16 + lm, colB);
        // hi * hi
        #pragma unroll
        for (int tm = 0; tm < 2; ++tm)
            #pragma unroll
            for (int tn = 0; tn < 4; ++tn)
                acc[tm][tn] = __builtin_amdgcn_wmma_f32_16x16x32_bf16(
                    false, ah[tm], false, b[tn], (short)0, acc[tm][tn], false, false);
        // lo * hi
        #pragma unroll
        for (int tm = 0; tm < 2; ++tm)
            al[tm] = load_fragA(&AsL[0][0], wrow * 32 + tm * 16 + lm, colA);
        #pragma unroll
        for (int tm = 0; tm < 2; ++tm)
            #pragma unroll
            for (int tn = 0; tn < 4; ++tn)
                acc[tm][tn] = __builtin_amdgcn_wmma_f32_16x16x32_bf16(
                    false, al[tm], false, b[tn], (short)0, acc[tm][tn], false, false);
        // hi * lo (reuse B fragment registers)
        #pragma unroll
        for (int tn = 0; tn < 4; ++tn)
            b[tn] = load_fragB(&BsL[0][0], wcol * 64 + tn * 16 + lm, colB);
        #pragma unroll
        for (int tm = 0; tm < 2; ++tm)
            #pragma unroll
            for (int tn = 0; tn < 4; ++tn)
                acc[tm][tn] = __builtin_amdgcn_wmma_f32_16x16x32_bf16(
                    false, ah[tm], false, b[tn], (short)0, acc[tm][tn], false, false);
    }

    // Epilogue: L2 -> 5-scale gaussian sum -> signed block accumulation
    const float c0 = cvals[0], c1 = cvals[1], c2 = cvals[2], c3 = cvals[3], c4 = cvals[4];
    float part = 0.0f;
    #pragma unroll
    for (int tm = 0; tm < 2; ++tm) {
        #pragma unroll
        for (int tn = 0; tn < 4; ++tn) {
            const int gj = j0 + wcol * 64 + tn * 16 + lm;
            const float sqj = sq[gj];
            #pragma unroll
            for (int v = 0; v < 8; ++v) {
                // C layout: VGPR v holds M=v (lanes 0-15) / M=v+8 (lanes 16-31)
                const int gi = i0 + wrow * 32 + tm * 16 + v + ((lane < 16) ? 0 : 8);
                const float g  = acc[tm][tn][v];
                const float L2 = fmaxf(sq[gi] + sqj - 2.0f * g, 0.0f);
                const float ks = __expf(-L2 * c0) + __expf(-L2 * c1) + __expf(-L2 * c2)
                               + __expf(-L2 * c3) + __expf(-L2 * c4);
                const float sgn = ((gi ^ gj) & HALF_B) ? -1.0f : 1.0f;  // XX/YY + , XY/YX -
                part += sgn * ks;
            }
        }
    }
    #pragma unroll
    for (int off = 16; off > 0; off >>= 1) part += __shfl_xor(part, off, 32);
    if (lane == 0) atomicAdd(&blk_acc, part);        // ds_add_f32
    __syncthreads();
    if (tid == 0) atomicAdd(signed_acc, blk_acc);    // global_atomic_add_f32
}

// ---------------- kernel 4: finalize scalar ---------------------------------
__global__ void mmd_final_kernel(const float* __restrict__ signed_acc, float* __restrict__ out) {
    out[0] = signed_acc[0] * (1.0f / ((float)HALF_B * (float)HALF_B));
}

extern "C" void kernel_launch(void* const* d_in, const int* in_sizes, int n_in,
                              void* d_out, int out_size, void* d_ws, size_t ws_size,
                              hipStream_t stream) {
    const float* src = (const float*)d_in[0];
    const float* tgt = (const float*)d_in[1];
    float* ws     = (float*)d_ws;
    float* sumsq  = ws + 0;
    float* sacc   = ws + 1;
    float* cvals  = ws + 2;     // 5 floats
    float* colsum = ws + 8;     // 256 floats
    float* sq     = ws + 264;   // 8192 floats
    __bf16* thi   = (__bf16*)(ws + 8704);              // 8192*256 bf16 (4 MB), 16B aligned
    __bf16* tlo   = thi + (size_t)N_TOT * DIMS;        // 8192*256 bf16 (4 MB)
    float* out    = (float*)d_out;

    mmd_init_kernel<<<1, 256, 0, stream>>>(ws);
    mmd_stats_kernel<<<32, 256, 0, stream>>>(src, tgt, sq, colsum, sumsq, thi, tlo);
    mmd_bw_kernel<<<1, 1, 0, stream>>>(sumsq, colsum, cvals);
    dim3 grid(N_TOT / BLK, N_TOT / BLK);   // 64 x 64 macro-tiles
    mmd_gram_kernel<<<grid, 256, 0, stream>>>(thi, tlo, sq, cvals, sacc);
    mmd_final_kernel<<<1, 1, 0, stream>>>(sacc, out);
}
